// GroupedNoShareTrans_15908558865272
// MI455X (gfx1250) — compile-verified, matrix operand order checked by reading, not run on hardware
//
#include <hip/hip_runtime.h>

// ---------------------------------------------------------------------------
// GroupedNoShareTrans fused inference for MI455X (gfx1250, wave32, WMMA)
// Stage 1: per-(sensor, 8-batch tile) fully fused 2-layer encoder in LDS.
// Stage 2: pooling + pool-linear (WMMA) + 16-token cross layer + final FC.
// All GEMMs: v_wmma_f32_16x16x32_f16 (f16 in, f32 accum).
// Fragment loads vectorized to ds_load_b128 (two contiguous 8-half runs per
// lane in the documented CDNA5 A/B layouts).
// ---------------------------------------------------------------------------

typedef _Float16 v16h __attribute__((ext_vector_type(16)));
typedef _Float16 h8   __attribute__((ext_vector_type(8)));
typedef float    v8f  __attribute__((ext_vector_type(8)));
typedef float    f4   __attribute__((ext_vector_type(4)));

#define NBATCH   512
#define NTIME    30
#define NSENS    64
#define NDIM     32
#define NHEAD    4
#define HDIM     8
#define NFF      128
#define NCOMP    16

// ---------------- LDS layout (dynamic shared; ~186 KB) ----------------------
struct SLds {
  float    hfp[256 * 32];       // fp32 hidden state (residual/LN precision)
  float    sc [8 * 32 * 32];    // per-wave score scratch (fp32)
  float    bi[96], bo[32], b1[128], b2[32];
  float    g1[32], be1[32], g2[32], be2[32];
  _Float16 buf [256 * 128];     // qkv (96 cols) / ff1 out (128 cols)
  _Float16 abuf[256 * 32];      // attention output
  _Float16 sh  [8 * 32 * 32];   // per-wave softmax probs (f16, padded w/ 0)
  _Float16 wi[96 * 32];         // qkv weight  [3D][D]
  _Float16 wo[32 * 32];         // out-proj    [D][D]   (also pool_w in stage2)
  _Float16 w1[128 * 32];        // ff1         [FF][D]
  _Float16 w2[32 * 128];        // ff2         [D][FF]
};

// ---------------- WMMA fragment builders (documented CDNA5 layouts) ---------
// A (16x32 f16): lane%16 = M row; lane holds K = lh*8+0..7 and 16+lh*8+0..7,
// i.e. two contiguous 8-half (16 B) runs -> two ds_load_b128.
// B mirrored: lane%16 = N col; same K striping.
// C/D (16x16 f32): lane%16 = N; VGPR j holds M = j + 8*(lane/16).

__device__ __forceinline__ v16h combine8(h8 lo, h8 hi) {
  return __builtin_shufflevector(lo, hi, 0, 1, 2, 3, 4, 5, 6, 7,
                                 8, 9, 10, 11, 12, 13, 14, 15);
}

__device__ __forceinline__ v16h ld_frag_f16(const _Float16* base, int ld, int lane) {
  const int m = lane & 15, lh = lane >> 4;
  const _Float16* p = base + m * ld + lh * 8;
  const h8 lo = *(const h8*)(p);
  const h8 hi = *(const h8*)(p + 16);
  return combine8(lo, hi);
}

// K-padded Q/K fragment: valid K dim = 8 (head dim). Only the lh==0 chunk of
// the first K run is ever non-zero; rows >= mmax are zeroed.
__device__ __forceinline__ v16h ld_frag_qk(const _Float16* base, int ld,
                                           int mmax, int lane) {
  const int m = lane & 15, lh = lane >> 4;
  h8 z = {};
  h8 lo = z;
  if (lh == 0 && m < mmax) lo = *(const h8*)(base + m * ld);
  return combine8(lo, z);
}

__device__ __forceinline__ v16h ld_frag_f32(const float* base, int ld, int lane) {
  const int m = lane & 15, lh = lane >> 4;
  const float* p = base + m * ld + lh * 8;
  const f4 x0 = *(const f4*)(p);
  const f4 x1 = *(const f4*)(p + 4);
  const f4 x2 = *(const f4*)(p + 16);
  const f4 x3 = *(const f4*)(p + 20);
  v16h a;
#pragma unroll
  for (int i = 0; i < 4; ++i) {
    a[i]      = (_Float16)x0[i];
    a[4 + i]  = (_Float16)x1[i];
    a[8 + i]  = (_Float16)x2[i];
    a[12 + i] = (_Float16)x3[i];
  }
  return a;
}

// B from K-major storage ([K][N] row-major), masked in K (V loads only).
__device__ __forceinline__ v16h ld_fragB_kmajor(const _Float16* base, int ld,
                                                int kmax, int lane) {
  const int n = lane & 15, lh = lane >> 4;
  v16h a;
#pragma unroll
  for (int j = 0; j < 8; ++j) {
    const int k0 = ((j >> 2) << 4) + (lh << 3) + ((j & 3) << 1);
    a[2 * j]     = (k0     < kmax) ? base[k0 * ld + n]       : (_Float16)0.f;
    a[2 * j + 1] = (k0 + 1 < kmax) ? base[(k0 + 1) * ld + n] : (_Float16)0.f;
  }
  return a;
}

__device__ __forceinline__ v8f wmma32(v16h a, v16h b, v8f c) {
  return __builtin_amdgcn_wmma_f32_16x16x32_f16(false, a, false, b, (short)0, c,
                                                false, false);
}

// ---------------- cooperative weight loads ----------------------------------
__device__ __forceinline__ void cp_f16(_Float16* dst, const float* src, int n, int tid) {
  for (int i = tid; i < n; i += 256) dst[i] = (_Float16)src[i];
}
__device__ __forceinline__ void cp_f32(float* dst, const float* src, int n, int tid) {
  for (int i = tid; i < n; i += 256) dst[i] = src[i];
}

// ---------------- per-row LayerNorm on hfp ----------------------------------
__device__ __forceinline__ void layer_norm_rows(SLds& L, int R, const float* g,
                                                const float* be, int tid) {
  if (tid < R) {
    float* row = &L.hfp[tid * NDIM];
    float m = 0.f;
#pragma unroll
    for (int d = 0; d < NDIM; ++d) m += row[d];
    m *= (1.f / NDIM);
    float v = 0.f;
#pragma unroll
    for (int d = 0; d < NDIM; ++d) { const float df = row[d] - m; v += df * df; }
    v *= (1.f / NDIM);
    const float inv = rsqrtf(v + 1e-5f);
#pragma unroll
    for (int d = 0; d < NDIM; ++d) row[d] = (row[d] - m) * inv * g[d] + be[d];
  }
}

// ---------------- one post-norm transformer encoder layer --------------------
// Operates fully in LDS on R = TOK*NB rows of hfp (fp32).
template <int TOK, int NB>
__device__ void encoder_layer(SLds& L, int tid) {
  constexpr int R  = TOK * NB;
  constexpr int MT = R / 16;               // row tiles
  constexpr int NT = (TOK + 15) / 16;      // token tiles (attention)
  static_assert(R % 16 == 0 && R <= 256, "bad tiling");
  const int wave = tid >> 5, lane = tid & 31;
  const int ln = lane & 15, lh = lane >> 4;

  // ---- QKV: [R,32] x [32,96] -> buf[:,0:96] (f16) --------------------------
  for (int t = wave; t < MT * 6; t += 8) {
    const int mt = t / 6, nt = t % 6;
    v16h a = ld_frag_f32(&L.hfp[mt * 16 * NDIM], NDIM, lane);
    v16h b = ld_frag_f16(&L.wi[nt * 16 * NDIM], NDIM, lane);
    v8f c = {};
    c = wmma32(a, b, c);
    const int n = nt * 16 + ln;
#pragma unroll
    for (int j = 0; j < 8; ++j) {
      const int r = mt * 16 + j + 8 * lh;
      L.buf[r * 128 + n] = (_Float16)(c[j] + L.bi[n]);
    }
  }
  __syncthreads();

  // ---- attention: per (batch, head) pair -----------------------------------
  const float scale = 0.35355339059327373f;  // 1/sqrt(8)
  float*    scw = &L.sc[wave * 1024];
  _Float16* shw = &L.sh[wave * 1024];
  for (int pair = wave; pair < NB * NHEAD; pair += 8) {
    const int bb = pair >> 2, hh = pair & 3;
    const int rb = bb * TOK;
    // scores = Q K^T  (K dim = 8, zero-padded to 32)
    for (int mt = 0; mt < NT; ++mt)
      for (int nt = 0; nt < NT; ++nt) {
        v16h a = ld_frag_qk(&L.buf[(rb + mt * 16) * 128 + hh * HDIM],
                            128, TOK - mt * 16, lane);
        v16h b = ld_frag_qk(&L.buf[(rb + nt * 16) * 128 + NDIM + hh * HDIM],
                            128, TOK - nt * 16, lane);
        v8f c = {};
        c = wmma32(a, b, c);
        const int col = nt * 16 + ln;
#pragma unroll
        for (int j = 0; j < 8; ++j)
          scw[(mt * 16 + j + 8 * lh) * 32 + col] = c[j];
      }
    // softmax: one row per lane; pad cols/rows with exact zeros for next GEMM
    {
      const int i = lane;
      if (i < TOK) {
        float mx = -1e30f;
        for (int j = 0; j < TOK; ++j) mx = fmaxf(mx, scw[i * 32 + j]);
        float sum = 0.f;
        for (int j = 0; j < TOK; ++j) sum += __expf((scw[i * 32 + j] - mx) * scale);
        const float inv = 1.f / sum;
        for (int j = 0; j < 32; ++j)
          shw[i * 32 + j] = (j < TOK)
              ? (_Float16)(__expf((scw[i * 32 + j] - mx) * scale) * inv)
              : (_Float16)0.f;
      } else {
        for (int j = 0; j < 32; ++j) shw[i * 32 + j] = (_Float16)0.f;
      }
    }
    // O = P V  (K dim = TOK, padded probs are zero)
    for (int mt = 0; mt < NT; ++mt) {
      v16h a = ld_frag_f16(&shw[mt * 16 * 32], 32, lane);
      v16h b = ld_fragB_kmajor(&L.buf[rb * 128 + 2 * NDIM + hh * HDIM], 128, TOK, lane);
      v8f c = {};
      c = wmma32(a, b, c);
      if (ln < HDIM) {
#pragma unroll
        for (int j = 0; j < 8; ++j) {
          const int tt = mt * 16 + j + 8 * lh;
          if (tt < TOK)
            L.abuf[(rb + tt) * NDIM + hh * HDIM + ln] = (_Float16)c[j];
        }
      }
    }
  }
  __syncthreads();

  // ---- out-proj + residual -> hfp ------------------------------------------
  for (int t = wave; t < MT * 2; t += 8) {
    const int mt = t >> 1, nt = t & 1;
    v16h a = ld_frag_f16(&L.abuf[mt * 16 * NDIM], NDIM, lane);
    v16h b = ld_frag_f16(&L.wo[nt * 16 * NDIM], NDIM, lane);
    v8f c = {};
    c = wmma32(a, b, c);
    const int n = nt * 16 + ln;
#pragma unroll
    for (int j = 0; j < 8; ++j) {
      const int r = mt * 16 + j + 8 * lh;
      L.hfp[r * NDIM + n] += c[j] + L.bo[n];
    }
  }
  __syncthreads();
  layer_norm_rows(L, R, L.g1, L.be1, tid);
  __syncthreads();

  // ---- FF1: [R,32] x [32,128], bias + ReLU -> buf --------------------------
  for (int t = wave; t < MT * 8; t += 8) {
    const int mt = t >> 3, nt = t & 7;
    v16h a = ld_frag_f32(&L.hfp[mt * 16 * NDIM], NDIM, lane);
    v16h b = ld_frag_f16(&L.w1[nt * 16 * NDIM], NDIM, lane);
    v8f c = {};
    c = wmma32(a, b, c);
    const int n = nt * 16 + ln;
#pragma unroll
    for (int j = 0; j < 8; ++j) {
      const int r = mt * 16 + j + 8 * lh;
      L.buf[r * 128 + n] = (_Float16)fmaxf(c[j] + L.b1[n], 0.f);
    }
  }
  __syncthreads();

  // ---- FF2: [R,128] x [128,32] (4 K steps) + residual -> hfp ---------------
  for (int t = wave; t < MT * 2; t += 8) {
    const int mt = t >> 1, nt = t & 1;
    v8f c = {};
#pragma unroll
    for (int ks = 0; ks < 4; ++ks) {
      v16h a = ld_frag_f16(&L.buf[mt * 16 * 128 + ks * 32], 128, lane);
      v16h b = ld_frag_f16(&L.w2[nt * 16 * 128 + ks * 32], 128, lane);
      c = wmma32(a, b, c);
    }
    const int n = nt * 16 + ln;
#pragma unroll
    for (int j = 0; j < 8; ++j) {
      const int r = mt * 16 + j + 8 * lh;
      L.hfp[r * NDIM + n] += c[j] + L.b2[n];
    }
  }
  __syncthreads();
  layer_norm_rows(L, R, L.g2, L.be2, tid);
  __syncthreads();
}

// ---------------- Stage 1: per-sensor fused 2-layer encoder ------------------
__global__ __launch_bounds__(256) void sensor_encoder_kernel(
    const float* __restrict__ x, const float* __restrict__ proj_w,
    const float* __restrict__ proj_b, const float* __restrict__ pos,
    const float* __restrict__ in_w, const float* __restrict__ in_b,
    const float* __restrict__ out_w, const float* __restrict__ out_b,
    const float* __restrict__ ln1w, const float* __restrict__ ln1b,
    const float* __restrict__ f1w, const float* __restrict__ f1b,
    const float* __restrict__ f2w, const float* __restrict__ f2b,
    const float* __restrict__ ln2w, const float* __restrict__ ln2b,
    float* __restrict__ reprs) {
  extern __shared__ char smem[];
  SLds& L = *reinterpret_cast<SLds*>(smem);
  const int tid = threadIdx.x;
  const int s  = blockIdx.x % NSENS;
  const int b0 = (blockIdx.x / NSENS) * 8;

  // embed: h = x[b,t,s] * proj_w[s] + proj_b[s] + pos[t]
  for (int i = tid; i < 8 * NTIME * NDIM; i += 256) {
    const int r = i >> 5, d = i & 31;
    const int bl = r / NTIME, t = r - bl * NTIME;
    const float xv = x[((size_t)(b0 + bl) * NTIME + t) * NSENS + s];
    L.hfp[i] = xv * proj_w[s * NDIM + d] + proj_b[s * NDIM + d] + pos[t * NDIM + d];
  }

  for (int l = 0; l < 2; ++l) {
    const int w = s * 2 + l;
    cp_f16(L.wi, in_w + (size_t)w * 96 * 32, 96 * 32, tid);
    cp_f32(L.bi, in_b + (size_t)w * 96, 96, tid);
    cp_f16(L.wo, out_w + (size_t)w * 32 * 32, 32 * 32, tid);
    cp_f32(L.bo, out_b + (size_t)w * 32, 32, tid);
    cp_f16(L.w1, f1w + (size_t)w * 128 * 32, 128 * 32, tid);
    cp_f32(L.b1, f1b + (size_t)w * 128, 128, tid);
    cp_f16(L.w2, f2w + (size_t)w * 32 * 128, 32 * 128, tid);
    cp_f32(L.b2, f2b + (size_t)w * 32, 32, tid);
    cp_f32(L.g1, ln1w + (size_t)w * 32, 32, tid);
    cp_f32(L.be1, ln1b + (size_t)w * 32, 32, tid);
    cp_f32(L.g2, ln2w + (size_t)w * 32, 32, tid);
    cp_f32(L.be2, ln2b + (size_t)w * 32, 32, tid);
    __syncthreads();
    encoder_layer<NTIME, 8>(L, tid);
  }

  // last token -> reprs[s][b][d]
  {
    const int bl = tid >> 5, d = tid & 31;
    reprs[((size_t)s * NBATCH + b0 + bl) * NDIM + d] =
        L.hfp[(bl * NTIME + (NTIME - 1)) * NDIM + d];
  }
}

// ---------------- Stage 2: pooling + cross layer + final FC ------------------
__global__ __launch_bounds__(256) void cross_kernel(
    const float* __restrict__ reprs, const float* __restrict__ pool_w,
    const float* __restrict__ pool_b, const float* __restrict__ cemb,
    const float* __restrict__ ciw, const float* __restrict__ cib,
    const float* __restrict__ cow, const float* __restrict__ cob,
    const float* __restrict__ cl1w, const float* __restrict__ cl1b,
    const float* __restrict__ cf1w, const float* __restrict__ cf1b,
    const float* __restrict__ cf2w, const float* __restrict__ cf2b,
    const float* __restrict__ cl2w, const float* __restrict__ cl2b,
    const float* __restrict__ fc_w, const float* __restrict__ fc_b,
    float* __restrict__ out) {
  extern __shared__ char smem[];
  SLds& L = *reinterpret_cast<SLds*>(smem);
  const int tid = threadIdx.x;
  const int b0 = blockIdx.x * 16;
  const int wave = tid >> 5, lane = tid & 31;
  const int ln = lane & 15, lh = lane >> 4;

  cp_f16(L.wo, pool_w, 32 * 32, tid);        // pool weight (reused slot)
  cp_f32(L.bo, pool_b, 32, tid);
  cp_f32(L.sc, cemb, NCOMP * NDIM, tid);     // stash cemb in score scratch

  // comp[lb][nc][k] = mean over 4 sensors of reprs  (one row per thread)
  {
    const int lb = tid >> 4, nc = tid & 15;
    const int b = b0 + lb;
#pragma unroll
    for (int k = 0; k < NDIM; ++k) {
      float acc = 0.f;
#pragma unroll
      for (int j = 0; j < 4; ++j)
        acc += reprs[((size_t)(nc * 4 + j) * NBATCH + b) * NDIM + k];
      L.abuf[tid * NDIM + k] = (_Float16)(acc * 0.25f);
    }
  }
  __syncthreads();

  // pool linear (WMMA) + bias + cemb -> hfp
  for (int t = wave; t < 32; t += 8) {
    const int mt = t >> 1, nt = t & 1;
    v16h a = ld_frag_f16(&L.abuf[mt * 16 * NDIM], NDIM, lane);
    v16h b = ld_frag_f16(&L.wo[nt * 16 * NDIM], NDIM, lane);
    v8f c = {};
    c = wmma32(a, b, c);
    const int n = nt * 16 + ln;
#pragma unroll
    for (int j = 0; j < 8; ++j) {
      const int r = mt * 16 + j + 8 * lh;
      L.hfp[r * NDIM + n] = c[j] + L.bo[n] + L.sc[(r & 15) * NDIM + n];
    }
  }
  __syncthreads();

  // cross-layer weights
  cp_f16(L.wi, ciw, 96 * 32, tid);  cp_f32(L.bi, cib, 96, tid);
  cp_f16(L.wo, cow, 32 * 32, tid);  cp_f32(L.bo, cob, 32, tid);
  cp_f16(L.w1, cf1w, 128 * 32, tid); cp_f32(L.b1, cf1b, 128, tid);
  cp_f16(L.w2, cf2w, 32 * 128, tid); cp_f32(L.b2, cf2b, 32, tid);
  cp_f32(L.g1, cl1w, 32, tid);  cp_f32(L.be1, cl1b, 32, tid);
  cp_f32(L.g2, cl2w, 32, tid);  cp_f32(L.be2, cl2b, 32, tid);
  __syncthreads();

  encoder_layer<NCOMP, 16>(L, tid);

  // final FC: out[b] = <s.flatten(), fc_w> + fc_b
  if (tid < 16) {
    float acc = fc_b[0];
    for (int i = 0; i < NCOMP * NDIM; ++i)
      acc += L.hfp[tid * NCOMP * NDIM + i] * fc_w[i];
    out[b0 + tid] = acc;
  }
}

// ---------------------------------------------------------------------------
extern "C" void kernel_launch(void* const* d_in, const int* in_sizes, int n_in,
                              void* d_out, int out_size, void* d_ws, size_t ws_size,
                              hipStream_t stream) {
  (void)in_sizes; (void)n_in; (void)out_size; (void)ws_size;
  const float* x       = (const float*)d_in[0];
  const float* proj_w  = (const float*)d_in[1];
  const float* proj_b  = (const float*)d_in[2];
  const float* pos     = (const float*)d_in[3];
  const float* in_w    = (const float*)d_in[4];
  const float* in_b    = (const float*)d_in[5];
  const float* out_w   = (const float*)d_in[6];
  const float* out_b   = (const float*)d_in[7];
  const float* ln1w    = (const float*)d_in[8];
  const float* ln1b    = (const float*)d_in[9];
  const float* f1w     = (const float*)d_in[10];
  const float* f1b     = (const float*)d_in[11];
  const float* f2w     = (const float*)d_in[12];
  const float* f2b     = (const float*)d_in[13];
  const float* ln2w    = (const float*)d_in[14];
  const float* ln2b    = (const float*)d_in[15];
  const float* pool_w  = (const float*)d_in[16];
  const float* pool_b  = (const float*)d_in[17];
  const float* cemb    = (const float*)d_in[18];
  const float* ciw     = (const float*)d_in[19];
  const float* cib     = (const float*)d_in[20];
  const float* cow     = (const float*)d_in[21];
  const float* cob     = (const float*)d_in[22];
  const float* cl1w    = (const float*)d_in[23];
  const float* cl1b    = (const float*)d_in[24];
  const float* cf1w    = (const float*)d_in[25];
  const float* cf1b    = (const float*)d_in[26];
  const float* cf2w    = (const float*)d_in[27];
  const float* cf2b    = (const float*)d_in[28];
  const float* cl2w    = (const float*)d_in[29];
  const float* cl2b    = (const float*)d_in[30];
  const float* fc_w    = (const float*)d_in[31];
  const float* fc_b    = (const float*)d_in[32];

  float* reprs = (float*)d_ws;  // [S][B][D] fp32 = 4 MB
  const size_t lds_bytes = sizeof(SLds);

  sensor_encoder_kernel<<<dim3(NSENS * (NBATCH / 8)), dim3(256), lds_bytes, stream>>>(
      x, proj_w, proj_b, pos, in_w, in_b, out_w, out_b, ln1w, ln1b,
      f1w, f1b, f2w, f2b, ln2w, ln2b, reprs);

  cross_kernel<<<dim3(NBATCH / 16), dim3(256), lds_bytes, stream>>>(
      reprs, pool_w, pool_b, cemb, ciw, cib, cow, cob, cl1w, cl1b,
      cf1w, cf1b, cf2w, cf2b, cl2w, cl2b, fc_w, fc_b, (float*)d_out);
}
